// Model5_6305011990978
// MI455X (gfx1250) — matrix operand order repaired and checked by reading, not compiled, not run on hardware
//
#include <hip/hip_runtime.h>
#include <hip/hip_bf16.h>

// GRU-D on MI455X (gfx1250): persistent batch-tile workgroups, bf16 WMMA
// 16x16x32 with f32 accumulation, recurrent state resident in LDS,
// weights converted once to bf16 and streamed from the 192MB L2 each step.

#define B_  512
#define T_  512
#define V_  64
#define H_  128
#define C_  4
#define G3  384   // 3*H
#define MT  16    // batch rows per workgroup (one WMMA M-tile)
#define NTW 3     // n-tiles per wave: 24 tiles / 8 waves

typedef __attribute__((ext_vector_type(16))) __bf16 v16bf;
typedef __attribute__((ext_vector_type(8)))  float  v8f;
typedef __attribute__((ext_vector_type(4)))  unsigned int u32x4;

union FragU { v16bf bf; u32x4 u[2]; };

__device__ __forceinline__ unsigned short f2bf(float f) {
  unsigned u = __float_as_uint(f);
  u += 0x7FFFu + ((u >> 16) & 1u);           // round-to-nearest-even
  return (unsigned short)(u >> 16);
}
__device__ __forceinline__ float bf2f(unsigned short h) {
  return __uint_as_float(((unsigned)h) << 16);
}
__device__ __forceinline__ float sigm(float x) { return 1.0f / (1.0f + __expf(-x)); }

// A-operand (16x32 bf16) from LDS, row-major [16][stride].
// lane<16: row=lane,    K = k0+0..7  and k0+16..23
// lane>=16: row=lane-16, K = k0+8..15 and k0+24..31
__device__ __forceinline__ v16bf load_a_lds(const unsigned short* base, int stride, int k0, int lane) {
  const int m = lane & 15;
  const int koff = k0 + ((lane & 16) >> 1);   // +8 for upper half-wave
  const unsigned short* p = base + m * stride + koff;
  FragU f;
  f.u[0] = *(const u32x4*)(p);
  f.u[1] = *(const u32x4*)(p + 16);
  return f.bf;
}

// B-operand (32x16 bf16) from global weight matrix W (row-major [384][ldk]),
// computing x @ W^T: B[k][n] = W[n][k].
// lane<16: col n = n0+lane,    K = k0+0..7  and k0+16..23
// lane>=16: col n = n0+lane-16, K = k0+8..15 and k0+24..31
__device__ __forceinline__ v16bf load_b_glb(const unsigned short* W, int ldk, int n0, int k0, int lane) {
  const int n = n0 + (lane & 15);
  const int koff = k0 + ((lane & 16) >> 1);   // +8 for upper half-wave
  const unsigned short* p = W + n * ldk + koff;
  FragU f;
  f.u[0] = *(const u32x4*)(p);
  f.u[1] = *(const u32x4*)(p + 16);
  return f.bf;
}

__device__ __forceinline__ v8f wmma_bf16(v16bf a, v16bf b, v8f c) {
  return __builtin_amdgcn_wmma_f32_16x16x32_bf16(false, a, false, b, (short)0, c, false, false);
}

// One GRU layer's gate pre-activations for a 16-row batch tile.
// r,z tiles (n0 < 256): one accumulator = b_ih+b_hh + x@Wih^T + h@Whh^T -> grz (bf16)
// n  tiles (n0 >= 256): gi = b_ih + x@Wih^T -> ginb ; gh = b_hh + h@Whh^T -> ghnb
template<int KX>
__device__ __forceinline__ void run_layer(
    const unsigned short* Ax, const unsigned short* Ah,
    const unsigned short* Wih, const unsigned short* Whh,
    const float* bih, const float* bhh,
    unsigned short (*grz)[2 * H_], unsigned short (*ginb)[H_], unsigned short (*ghnb)[H_],
    int wave, int lane)
{
  const int nl = lane & 15;
  const int mb = (lane & 16) ? 8 : 0;         // D layout: lanes>=16 hold M=8..15
  #pragma unroll
  for (int ti = 0; ti < NTW; ++ti) {
    const int n0 = (wave * NTW + ti) * 16;    // wave-uniform
    const int n = n0 + nl;
    const float bi = bih[n];
    const float bh = bhh[n];
    if (n0 < 2 * H_) {                        // r / z gates: fuse gi+gh
      v8f acc;
      const float bs = bi + bh;
      #pragma unroll
      for (int j = 0; j < 8; ++j) acc[j] = bs;
      #pragma unroll
      for (int kt = 0; kt < KX / 32; ++kt)
        acc = wmma_bf16(load_a_lds(Ax, KX, kt * 32, lane),
                        load_b_glb(Wih, KX, n0, kt * 32, lane), acc);
      #pragma unroll
      for (int kt = 0; kt < H_ / 32; ++kt)
        acc = wmma_bf16(load_a_lds(Ah, H_, kt * 32, lane),
                        load_b_glb(Whh, H_, n0, kt * 32, lane), acc);
      #pragma unroll
      for (int j = 0; j < 8; ++j) grz[mb + j][n] = f2bf(acc[j]);
    } else {                                  // n gate: keep gi / gh separate
      v8f ai, ah;
      #pragma unroll
      for (int j = 0; j < 8; ++j) { ai[j] = bi; ah[j] = bh; }
      #pragma unroll
      for (int kt = 0; kt < KX / 32; ++kt)
        ai = wmma_bf16(load_a_lds(Ax, KX, kt * 32, lane),
                       load_b_glb(Wih, KX, n0, kt * 32, lane), ai);
      #pragma unroll
      for (int kt = 0; kt < H_ / 32; ++kt)
        ah = wmma_bf16(load_a_lds(Ah, H_, kt * 32, lane),
                       load_b_glb(Whh, H_, n0, kt * 32, lane), ah);
      const int c = n - 2 * H_;
      #pragma unroll
      for (int j = 0; j < 8; ++j) { ginb[mb + j][c] = f2bf(ai[j]); ghnb[mb + j][c] = f2bf(ah[j]); }
    }
  }
}

__device__ __forceinline__ void gate_update(
    float (*hfl)[H_], unsigned short (*hbl)[H_],
    unsigned short (*grz)[2 * H_], unsigned short (*ginb)[H_], unsigned short (*ghnb)[H_],
    int tid)
{
  for (int idx = tid; idx < MT * H_; idx += 256) {
    const int m = idx >> 7, c = idx & (H_ - 1);
    const float r  = sigm(bf2f(grz[m][c]));
    const float z  = sigm(bf2f(grz[m][H_ + c]));
    const float nn = tanhf(bf2f(ginb[m][c]) + r * bf2f(ghnb[m][c]));
    const float hnew = (1.0f - z) * nn + z * hfl[m][c];
    hfl[m][c] = hnew;
    hbl[m][c] = f2bf(hnew);
  }
}

__global__ void cvt_bf16_kernel(const float* __restrict__ src, unsigned short* __restrict__ dst, int n) {
  const int i = blockIdx.x * blockDim.x + threadIdx.x;
  if (i < n) dst[i] = f2bf(src[i]);
}

__global__ __launch_bounds__(256) void grud_kernel(
    const float* __restrict__ x, const float* __restrict__ d, const float* __restrict__ m1,
    const float* __restrict__ h01, const float* __restrict__ h02, const float* __restrict__ h03,
    const unsigned short* __restrict__ Wih1, const unsigned short* __restrict__ Whh1,
    const float* __restrict__ bih1, const float* __restrict__ bhh1,
    const unsigned short* __restrict__ Wih2, const unsigned short* __restrict__ Whh2,
    const float* __restrict__ bih2, const float* __restrict__ bhh2,
    const unsigned short* __restrict__ Wih3, const unsigned short* __restrict__ Whh3,
    const float* __restrict__ bih3, const float* __restrict__ bhh3,
    const float* __restrict__ Wout, const float* __restrict__ bout,
    const float* __restrict__ Wdx, const float* __restrict__ bdx,
    const float* __restrict__ Wdh, const float* __restrict__ bdh,
    float* __restrict__ out)
{
  __shared__ __align__(16) float          x0f[MT][V_];        // 4 KB, baseline x0
  __shared__ __align__(16) unsigned short x1b[MT][V_];        // 2 KB, imputed input (bf16)
  __shared__ __align__(16) float          hf[3][MT][H_];      // 24 KB, fp32 master state
  __shared__ __align__(16) unsigned short hb[3][MT][H_];      // 12 KB, bf16 state for WMMA A
  __shared__ __align__(16) unsigned short grz[MT][2 * H_];    // 8 KB, r/z pre-activations
  __shared__ __align__(16) unsigned short ginb[MT][H_];       // 4 KB
  __shared__ __align__(16) unsigned short ghnb[MT][H_];       // 4 KB
  __shared__ float ylog[MT][C_];
  __shared__ float dld[MT];

  const int tid  = threadIdx.x;
  const int wave = tid >> 5;
  const int lane = tid & 31;
  const int b0   = blockIdx.x * MT;

  // ---- init: x0, h0 ----
  for (int idx = tid; idx < MT * V_; idx += 256) {
    const int m = idx >> 6, v = idx & (V_ - 1);
    const float xv = x[((size_t)(b0 + m) * T_) * V_ + v];
    x0f[m][v] = xv;
    x1b[m][v] = f2bf(xv);                     // t=0 uses raw x0
  }
  for (int idx = tid; idx < MT * H_; idx += 256) {
    const int m = idx >> 7, c = idx & (H_ - 1);
    const float a1 = h01[(size_t)(b0 + m) * H_ + c];
    const float a2 = h02[(size_t)(b0 + m) * H_ + c];
    const float a3 = h03[(size_t)(b0 + m) * H_ + c];
    hf[0][m][c] = a1; hb[0][m][c] = f2bf(a1);
    hf[1][m][c] = a2; hb[1][m][c] = f2bf(a2);
    hf[2][m][c] = a3; hb[2][m][c] = f2bf(a3);
  }

  const unsigned short* WihA[3] = {Wih1, Wih2, Wih3};
  const unsigned short* WhhA[3] = {Whh1, Whh2, Whh3};
  const float* bihA[3] = {bih1, bih2, bih3};
  const float* bhhA[3] = {bhh1, bhh2, bhh3};

  for (int t = 0; t < T_; ++t) {
    if (t > 0) {
      if (tid < MT) dld[tid] = d[(size_t)(b0 + tid) * T_ + t];
      __syncthreads();
      // hidden-state decay: h *= exp(-relu(dt*W_dh + b_dh))
      for (int idx = tid; idx < MT * H_; idx += 256) {
        const int m = idx >> 7, c = idx & (H_ - 1);
        const float g = __expf(-fmaxf(0.0f, dld[m] * Wdh[c] + bdh[c]));
        #pragma unroll
        for (int l = 0; l < 3; ++l) {
          const float hv = hf[l][m][c] * g;
          hf[l][m][c] = hv;
          hb[l][m][c] = f2bf(hv);
        }
      }
      // masked imputation of the input
      for (int idx = tid; idx < MT * V_; idx += 256) {
        const int m = idx >> 6, v = idx & (V_ - 1);
        const float gx = __expf(-fmaxf(0.0f, dld[m] * Wdx[v] + bdx[v]));
        const size_t base = ((size_t)(b0 + m) * T_ + t) * V_ + v;
        const float xt = x[base];
        const float xp = x[base - V_];
        const float mk = m1[base];
        const float xv = (xp * gx + (1.0f - gx) * x0f[m][v]) * (1.0f - mk) + mk * xt;
        x1b[m][v] = f2bf(xv);
      }
    }

    #pragma unroll
    for (int l = 0; l < 3; ++l) {
      __syncthreads();                        // A-operands + gate scratch ready/free
      if (l == 0)
        run_layer<V_>(&x1b[0][0],       &hb[0][0][0], WihA[0], WhhA[0], bihA[0], bhhA[0],
                      grz, ginb, ghnb, wave, lane);
      else
        run_layer<H_>(&hb[l - 1][0][0], &hb[l][0][0], WihA[l], WhhA[l], bihA[l], bhhA[l],
                      grz, ginb, ghnb, wave, lane);
      __syncthreads();                        // gate pre-activations visible
      gate_update(hf[l], hb[l], grz, ginb, ghnb, tid);
    }
    __syncthreads();                          // h3 final for this step

    // head: logits (16 x 4), then softmax over C=4
    if (tid < MT * C_) {
      const int m = tid >> 2, c = tid & 3;
      float acc = bout[c];
      #pragma unroll 8
      for (int k = 0; k < H_; ++k) acc += hf[2][m][k] * Wout[c * H_ + k];
      ylog[m][c] = acc;
    }
    __syncthreads();
    if (tid < MT) {
      const float l0 = ylog[tid][0], l1 = ylog[tid][1], l2 = ylog[tid][2], l3 = ylog[tid][3];
      const float mx = fmaxf(fmaxf(l0, l1), fmaxf(l2, l3));
      const float e0 = __expf(l0 - mx), e1 = __expf(l1 - mx);
      const float e2 = __expf(l2 - mx), e3 = __expf(l3 - mx);
      const float inv = 1.0f / (e0 + e1 + e2 + e3);
      float* o = out + ((size_t)(b0 + tid) * T_ + t) * C_;
      o[0] = e0 * inv; o[1] = e1 * inv; o[2] = e2 * inv; o[3] = e3 * inv;
    }
    __syncthreads();                          // protect hf/ylog before next step
  }
}

extern "C" void kernel_launch(void* const* d_in, const int* in_sizes, int n_in,
                              void* d_out, int out_size, void* d_ws, size_t ws_size,
                              hipStream_t stream) {
  (void)in_sizes; (void)n_in; (void)out_size; (void)ws_size;
  const float* x    = (const float*)d_in[0];
  const float* dd   = (const float*)d_in[1];
  const float* m1   = (const float*)d_in[2];
  const float* h01  = (const float*)d_in[3];
  const float* h02  = (const float*)d_in[4];
  const float* h03  = (const float*)d_in[5];
  const float* Wih1f = (const float*)d_in[6];
  const float* Whh1f = (const float*)d_in[7];
  const float* bih1 = (const float*)d_in[8];
  const float* bhh1 = (const float*)d_in[9];
  const float* Wih2f = (const float*)d_in[10];
  const float* Whh2f = (const float*)d_in[11];
  const float* bih2 = (const float*)d_in[12];
  const float* bhh2 = (const float*)d_in[13];
  const float* Wih3f = (const float*)d_in[14];
  const float* Whh3f = (const float*)d_in[15];
  const float* bih3 = (const float*)d_in[16];
  const float* bhh3 = (const float*)d_in[17];
  const float* Wout = (const float*)d_in[18];
  const float* bout = (const float*)d_in[19];
  const float* Wdx  = (const float*)d_in[20];
  const float* bdx  = (const float*)d_in[21];
  const float* Wdh  = (const float*)d_in[22];
  const float* bdh  = (const float*)d_in[23];

  // bf16 weight copies in workspace (540,672 bytes total)
  unsigned short* ws    = (unsigned short*)d_ws;
  unsigned short* Wih1b = ws;
  unsigned short* Whh1b = Wih1b + G3 * V_;
  unsigned short* Wih2b = Whh1b + G3 * H_;
  unsigned short* Whh2b = Wih2b + G3 * H_;
  unsigned short* Wih3b = Whh2b + G3 * H_;
  unsigned short* Whh3b = Wih3b + G3 * H_;

  cvt_bf16_kernel<<<(G3 * V_ + 255) / 256, 256, 0, stream>>>(Wih1f, Wih1b, G3 * V_);
  cvt_bf16_kernel<<<(G3 * H_ + 255) / 256, 256, 0, stream>>>(Whh1f, Whh1b, G3 * H_);
  cvt_bf16_kernel<<<(G3 * H_ + 255) / 256, 256, 0, stream>>>(Wih2f, Wih2b, G3 * H_);
  cvt_bf16_kernel<<<(G3 * H_ + 255) / 256, 256, 0, stream>>>(Whh2f, Whh2b, G3 * H_);
  cvt_bf16_kernel<<<(G3 * H_ + 255) / 256, 256, 0, stream>>>(Wih3f, Wih3b, G3 * H_);
  cvt_bf16_kernel<<<(G3 * H_ + 255) / 256, 256, 0, stream>>>(Whh3f, Whh3b, G3 * H_);

  grud_kernel<<<B_ / MT, 256, 0, stream>>>(
      x, dd, m1, h01, h02, h03,
      Wih1b, Whh1b, bih1, bhh1,
      Wih2b, Whh2b, bih2, bhh2,
      Wih3b, Whh3b, bih3, bhh3,
      Wout, bout, Wdx, bdx, Wdh, bdh,
      (float*)d_out);
}
